// E30DiagonalGatedCell_48481590837436
// MI455X (gfx1250) — matrix operand order repaired and checked by reading, not compiled, not run on hardware
//
#include <hip/hip_runtime.h>
#include <hip/hip_bf16.h>
#include <math.h>

// Problem sizes (fixed by the reference).
#define T_STEPS 2048
#define BB 16
#define DD 1024

#define SCAN_NWG 16     // persistent workgroups in the scan
#define SCAN_WAVES 4    // waves per scan WG (wave32) -> 64 waves = 64 column tiles

typedef __attribute__((ext_vector_type(16))) __bf16 bf16x16;
typedef __attribute__((ext_vector_type(8)))  float  f32x8;
typedef __attribute__((ext_vector_type(4)))  float  f32x4;
typedef __attribute__((ext_vector_type(4)))  unsigned int u32x4;

union BfPack  { bf16x16 v; u32x4 u[2]; };
union AccPack { f32x8 v; float f[8]; };

// ---------------------------------------------------------------------------
// Kernel 0: one-time prep.
//  - W_x, W_h (fp32, [e][d] row-major) -> bf16 transposed [k][e] so the WMMA
//    B operand (lane = k, 16 contiguous n) is a contiguous 32B load per lane.
//  - h[0] = h0 into the output h region; bf16 broadcast buffer hbf0 = bf16(h0).
//  - zero the grid-barrier words (deterministic across graph replays).
// ---------------------------------------------------------------------------
__global__ void prep_kernel(const float* __restrict__ Wx,
                            const float* __restrict__ Wh,
                            const float* __restrict__ h0,
                            unsigned short* __restrict__ WxT_,
                            unsigned short* __restrict__ WhT_,
                            unsigned short* __restrict__ hbf0_,
                            float* __restrict__ h_out,
                            unsigned* __restrict__ sync) {
  __bf16* WxT  = (__bf16*)WxT_;
  __bf16* WhT  = (__bf16*)WhT_;
  __bf16* hbf0 = (__bf16*)hbf0_;
  int idx = blockIdx.x * blockDim.x + threadIdx.x;
  if (idx < DD * DD) {
    int k = idx >> 10;        // idx / D
    int e = idx & (DD - 1);   // idx % D
    WxT[idx] = (__bf16)Wx[e * DD + k];   // WxT[k][e] = W_x[e][k]
    WhT[idx] = (__bf16)Wh[e * DD + k];   // WhT[k][j] = W_h[j][k]
  }
  if (idx < BB * DD) {
    float v = h0[idx];
    hbf0[idx]  = (__bf16)v;
    h_out[idx] = v;                      // h[0] = h0
  }
  if (idx == 0) { sync[0] = 0u; sync[1] = 0u; }
}

// ---------------------------------------------------------------------------
// Kernel 1: xp = X * W_x^T + b   (M=32768, N=1024, K=1024), bf16 WMMA.
// One wave computes a 16x64 output strip (4 N-tiles) so each A chunk feeds
// 4 wmma ops. Result written into the *output* region of d_out (reused as xp
// scratch; the scan overwrites it in place with the gated output).
// ---------------------------------------------------------------------------
__global__ void __launch_bounds__(256)
xp_gemm_kernel(const float* __restrict__ x,
               const unsigned short* __restrict__ WxT_,
               const float* __restrict__ bias,
               float* __restrict__ xp) {
  const __bf16* WxT = (const __bf16*)WxT_;
  const int lane  = threadIdx.x & 31;
  const int wave  = threadIdx.x >> 5;          // 0..7
  const int job   = blockIdx.x * 8 + wave;     // 0..32767
  const int mtile = job >> 4;                  // 0..2047
  const int m0    = mtile * 16;
  const int e0    = (job & 15) * 64;           // N group of 64 columns
  const int row   = lane & 15;                 // A row / C column index
  const int koff  = (lane & 16) ? 8 : 0;       // A K-half select per lane group
  const int moff  = (lane & 16) ? 8 : 0;       // C row offset per lane group

  f32x8 acc[4] = {};
  const float* xrow = x + (size_t)(m0 + row) * DD;

  for (int c = 0; c < 32; ++c) {
    const int kbase = c * 32;
    // ---- A operand: rows m0+row, K = kbase+koff..+7 and kbase+16+koff..+7
    BfPack a;
    {
      const f32x4* p0 = (const f32x4*)(xrow + kbase + koff);
      const f32x4* p1 = (const f32x4*)(xrow + kbase + 16 + koff);
      f32x4 f0 = p0[0], f1 = p0[1], f2 = p1[0], f3 = p1[1];
      float af[16] = {f0.x, f0.y, f0.z, f0.w, f1.x, f1.y, f1.z, f1.w,
                      f2.x, f2.y, f2.z, f2.w, f3.x, f3.y, f3.z, f3.w};
#pragma unroll
      for (int i = 0; i < 16; ++i) a.v[i] = (__bf16)af[i];
    }
    // ---- 4 B operands + wmma: lane holds column n = row, K = kbase+koff..+7 and kbase+16+koff..+7
    const __bf16* brow = WxT + (size_t)(kbase + koff) * DD + e0 + row;
#pragma unroll
    for (int nt = 0; nt < 4; ++nt) {
      BfPack b;
#pragma unroll
      for (int i = 0; i < 16; ++i)
        b.v[i] = brow[(size_t)(i < 8 ? i : i + 8) * DD + nt * 16];
      acc[nt] = __builtin_amdgcn_wmma_f32_16x16x32_bf16(
          false, a.v, false, b.v, (short)0, acc[nt], false, false);
    }
  }

  // ---- epilogue: add bias, store C (VGPR r -> row r+moff, col = row)
#pragma unroll
  for (int nt = 0; nt < 4; ++nt) {
    float bval = bias[e0 + nt * 16 + row];
    AccPack ap; ap.v = acc[nt];
#pragma unroll
    for (int r = 0; r < 8; ++r) {
      size_t m = (size_t)(m0 + r + moff);
      xp[m * DD + e0 + nt * 16 + row] = ap.f[r] + bval;
    }
  }
}

// ---------------------------------------------------------------------------
// Grid-wide generation barrier for the persistent scan kernel.
// ---------------------------------------------------------------------------
__device__ __forceinline__ void grid_barrier(unsigned* cnt, unsigned* gen,
                                             unsigned nwg) {
  __threadfence();        // publish this thread's stores to device scope
  __syncthreads();
  if (threadIdx.x == 0) {
    unsigned g = __hip_atomic_load(gen, __ATOMIC_RELAXED, __HIP_MEMORY_SCOPE_AGENT);
    if (__hip_atomic_fetch_add(cnt, 1u, __ATOMIC_ACQ_REL,
                               __HIP_MEMORY_SCOPE_AGENT) == nwg - 1u) {
      __hip_atomic_store(cnt, 0u, __ATOMIC_RELAXED, __HIP_MEMORY_SCOPE_AGENT);
      __hip_atomic_fetch_add(gen, 1u, __ATOMIC_RELEASE, __HIP_MEMORY_SCOPE_AGENT);
    } else {
      while (__hip_atomic_load(gen, __ATOMIC_ACQUIRE,
                               __HIP_MEMORY_SCOPE_AGENT) == g) {
        __builtin_amdgcn_s_sleep(1);
      }
    }
  }
  __syncthreads();
  // Acquire per-wave so every CU's L0 sees the other WGPs' h writes.
  (void)__hip_atomic_load(gen, __ATOMIC_ACQUIRE, __HIP_MEMORY_SCOPE_AGENT);
}

// ---------------------------------------------------------------------------
// Kernel 2: persistent recurrent scan.
// 16 WGs x 4 waves; wave w of WG g owns columns [ (g*4+w)*16 , +16 ).
// W_h^T slice (1024x16 bf16 per wave = 32KB, 128KB per WG) is staged into LDS
// with GLOBAL_LOAD_ASYNC_TO_LDS_B128 (ASYNCcnt) and stays resident.
// h broadcast: bf16 double buffer in global (L2-resident, 32KB).
// Per step per wave: 32 x v_wmma_f32_16x16x32_bf16 over K=1024, then
// tanh + SiLU gate epilogue, then grid barrier.
// ---------------------------------------------------------------------------
__global__ void __launch_bounds__(128)
scan_kernel(const float* __restrict__ z,
            const unsigned short* __restrict__ WhT_,
            const float* __restrict__ g_z,
            const float* __restrict__ g_h,
            const float* __restrict__ b_gate,
            float* __restrict__ out_xp,          // xp in / gated out (in place)
            float* __restrict__ h_out,           // [T+1,B,D]
            unsigned short* __restrict__ hbf0_,
            unsigned short* __restrict__ hbf1_,
            unsigned* __restrict__ sync) {
  extern __shared__ __align__(16) char smem_raw[];
  __bf16* smem = (__bf16*)smem_raw;              // [SCAN_WAVES][1024][16]
  const __bf16* WhT = (const __bf16*)WhT_;
  __bf16* hbf0 = (__bf16*)hbf0_;
  __bf16* hbf1 = (__bf16*)hbf1_;

  const int lane = threadIdx.x & 31;
  const int wave = threadIdx.x >> 5;                   // 0..3
  const int j0   = (blockIdx.x * SCAN_WAVES + wave) * 16;  // output column base
  const int row  = lane & 15;
  const int koff = (lane & 16) ? 8 : 0;
  const int moff = (lane & 16) ? 8 : 0;

  // ---- async-stage W_h^T[:, j0:j0+16] into LDS: smem[wave][k][n].
  // Per lane: one 32-byte LDS row (k = lane + 32*i), copied as two async B128
  // transfers. INST_OFFSET applies to both the global and LDS addresses, so
  // the upper 16B half reuses the same operands with offset:16.
  // Dynamic LDS starts at offset 0 (no static __shared__ in this kernel).
  {
    const unsigned lds_base = (unsigned)wave * (DD * 16 * 2); // bytes
    for (int k = lane; k < DD; k += 32) {
      unsigned long long gaddr =
          (unsigned long long)(const void*)(WhT + (size_t)k * DD + j0);
      unsigned laddr = lds_base + (unsigned)k * 32;
      asm volatile("global_load_async_to_lds_b128 %0, %1, off"
                   :: "v"(laddr), "v"(gaddr) : "memory");
      asm volatile("global_load_async_to_lds_b128 %0, %1, off offset:16"
                   :: "v"(laddr), "v"(gaddr) : "memory");
    }
  }
  asm volatile("s_wait_asynccnt 0x0" ::: "memory");
  __syncthreads();

  // per-lane gate parameters (column = j0 + row for this lane's C elements)
  const float gz = g_z[j0 + row];
  const float gh = g_h[j0 + row];
  const float bg = b_gate[j0 + row];
  const __bf16* bbase = smem + (size_t)wave * (DD * 16);

  for (int t = 0; t < T_STEPS; ++t) {
    const __bf16* hsrc = (t & 1) ? hbf1 : hbf0;
    __bf16*       hdst = (t & 1) ? hbf0 : hbf1;

    // ---- 16x16 tile of h_prev @ W_h^T, K = 1024
    f32x8 acc = {};
    const __bf16* hrow = hsrc + row * DD;
#pragma unroll 4
    for (int c = 0; c < 32; ++c) {
      BfPack a, b;
      a.u[0] = *(const u32x4*)(hrow + c * 32 + koff);
      a.u[1] = *(const u32x4*)(hrow + c * 32 + 16 + koff);
#pragma unroll
      for (int i = 0; i < 16; ++i)
        b.v[i] = bbase[(size_t)(c * 32 + koff + (i < 8 ? i : i + 8)) * 16 + row];
      acc = __builtin_amdgcn_wmma_f32_16x16x32_bf16(
          false, a.v, false, b.v, (short)0, acc, false, false);
    }

    // ---- epilogue: h_new = tanh(xp + acc); out = h_new * silu(gate pre-act)
    const size_t tb = (size_t)t * (BB * DD);
    AccPack ap; ap.v = acc;
#pragma unroll
    for (int r = 0; r < 8; ++r) {
      const int m = r + moff;
      const size_t idx = tb + (size_t)m * DD + j0 + row;
      float hv = tanhf(out_xp[idx] + ap.f[r]);
      h_out[tb + (size_t)(BB * DD) + (size_t)m * DD + j0 + row] = hv; // h[t+1]
      hdst[m * DD + j0 + row] = (__bf16)hv;                           // bf16 bcast
      float zt = z[idx];
      float pre = zt * gz + hv * gh + bg;
      float gate = pre / (1.0f + __expf(-pre));                       // silu
      out_xp[idx] = hv * gate;                                        // overwrite xp
    }

    // hint the next timestep's xp/z into cache while we wait at the barrier
    if (t + 1 < T_STEPS) {
      const size_t nidx = tb + (size_t)(BB * DD) + (size_t)moff * DD + j0 + row;
      __builtin_prefetch(out_xp + nidx, 0, 0);
      __builtin_prefetch(z + nidx, 0, 0);
    }

    grid_barrier(&sync[0], &sync[1], SCAN_NWG);
  }
}

// ---------------------------------------------------------------------------
// Host entry.
// d_out layout: [0, T*B*D)           = output (first used as xp scratch)
//               [T*B*D, +(T+1)*B*D)  = h
// d_ws layout:  WxT bf16 (2MB) | WhT bf16 (2MB) | hbf0 (32KB) | hbf1 (32KB)
//               | sync (2 u32)   -> ~4.26 MB total.
// ---------------------------------------------------------------------------
extern "C" void kernel_launch(void* const* d_in, const int* in_sizes, int n_in,
                              void* d_out, int out_size, void* d_ws, size_t ws_size,
                              hipStream_t stream) {
  const float* x   = (const float*)d_in[0];
  const float* z   = (const float*)d_in[1];
  const float* h0  = (const float*)d_in[2];
  const float* Wx  = (const float*)d_in[3];
  const float* Wh  = (const float*)d_in[4];
  const float* b   = (const float*)d_in[5];
  const float* g_z = (const float*)d_in[6];
  const float* g_h = (const float*)d_in[7];
  const float* b_g = (const float*)d_in[8];

  float* out  = (float*)d_out;                         // [T,B,D]
  float* hout = out + (size_t)T_STEPS * BB * DD;       // [T+1,B,D]

  char* ws = (char*)d_ws;
  unsigned short* WxT  = (unsigned short*)ws;                                  // 2 MB
  unsigned short* WhT  = (unsigned short*)(ws + (size_t)DD * DD * 2);          // 2 MB
  unsigned short* hbf0 = (unsigned short*)(ws + (size_t)DD * DD * 4);          // 32 KB
  unsigned short* hbf1 = hbf0 + (size_t)BB * DD;                               // 32 KB
  unsigned*       sync = (unsigned*)(hbf1 + (size_t)BB * DD);

  // Kernel 0: weight transpose+cast, h[0] copy, barrier reset.
  prep_kernel<<<(DD * DD + 255) / 256, 256, 0, stream>>>(
      Wx, Wh, h0, WxT, WhT, hbf0, hout, sync);

  // Kernel 1: xp = X * W_x^T + b  (into the output region, overwritten later).
  xp_gemm_kernel<<<4096, 256, 0, stream>>>(x, WxT, b, out);

  // Kernel 2: persistent recurrent scan (128 KB dynamic LDS per WG).
  const size_t lds_bytes = (size_t)SCAN_WAVES * DD * 16 * sizeof(unsigned short);
  hipFuncSetAttribute(reinterpret_cast<const void*>(scan_kernel),
                      hipFuncAttributeMaxDynamicSharedMemorySize,
                      (int)lds_bytes);
  scan_kernel<<<SCAN_NWG, 128, lds_bytes, stream>>>(
      z, WhT, g_z, g_h, b_g, out, hout, hbf0, hbf1, sync);
}
